// SelfAttention_65481071397187
// MI455X (gfx1250) — matrix operand order repaired and checked by reading, not compiled, not run on hardware
//
#include <hip/hip_runtime.h>
#include <hip/hip_bf16.h>

// ---------------------------------------------------------------------------
// Self-attention (B=4, C=128, H=W=64) — flash-attention style, bf16 WMMA.
//   q = (Wq x + bq) * log2(e)/sqrt(C)  -> bf16 [b][n][32] (d=16 padded to 32)
//   k = (Wk x + bk)                    -> bf16 [b][n][32]
//   v = (Wv x + bv)                    -> bf16 [b][c][N]
//   out = gamma * (V softmax(qT k)) + x,  streamed (no NxN matrix in HBM)
// Two waves per workgroup split the m-range (flash split-K) and merge their
// partial (max, sum, acc) states through LDS — doubles waves/SIMD so loads and
// WMMA hazards are hidden by the other wave. Scores live in the log2 domain;
// row sums are computed with WMMA (P x ones); K tiles double-buffered.
// ---------------------------------------------------------------------------

typedef __bf16 bf16;
typedef __attribute__((ext_vector_type(16))) __bf16 v16bf;
typedef __attribute__((ext_vector_type(8)))  __bf16 v8bf;
typedef __attribute__((ext_vector_type(8)))  float  v8f;

constexpr int B_DIM = 4;
constexpr int C_DIM = 128;
constexpr int N_PIX = 4096;   // 64*64
constexpr int MT    = 64;     // m-tile width per flash iteration
constexpr int MHALF = N_PIX / 2;   // m-range per wave

// -------------------------------- projections ------------------------------

// q and k: 2 * B * N * 16 threads. Writes [b][n][32] bf16, upper 16 zeroed.
__global__ __launch_bounds__(256)
void proj_qk_kernel(const float* __restrict__ x,
                    const float* __restrict__ wq, const float* __restrict__ bq,
                    const float* __restrict__ wk, const float* __restrict__ bk,
                    bf16* __restrict__ qT, bf16* __restrict__ kT)
{
    int idx   = blockIdx.x * blockDim.x + threadIdx.x;   // < 2*2^18
    int which = idx >> 18;                               // 0 = q, 1 = k
    int r     = idx & ((1 << 18) - 1);
    int d     = r & 15;
    int n     = (r >> 4) & (N_PIX - 1);
    int b     = r >> 16;

    const float* w  = which ? wk : wq;
    const float* bs = which ? bk : bq;
    bf16*        o  = which ? kT : qT;

    const float* xp = x + (size_t)b * C_DIM * N_PIX + n;
    const float* wp = w + d * C_DIM;
    float s = 0.f;
#pragma unroll 8
    for (int c = 0; c < C_DIM; ++c)
        s = fmaf(xp[(size_t)c * N_PIX], wp[c], s);
    s += bs[d];
    // fold 1/sqrt(C) and log2(e) into q: scores land in the log2 domain
    if (!which) s *= (1.4426950408889634f * 0.08838834764831845f);

    size_t ob = ((size_t)b * N_PIX + n) * 32;
    o[ob + d]      = (bf16)s;
    o[ob + 16 + d] = (bf16)0.f;                          // K padding
}

// v: B * C * N threads. Writes [b][c][N] bf16 (natural layout).
__global__ __launch_bounds__(256)
void proj_v_kernel(const float* __restrict__ x,
                   const float* __restrict__ wv, const float* __restrict__ bv,
                   bf16* __restrict__ vout)
{
    int idx = blockIdx.x * blockDim.x + threadIdx.x;     // < 2^21
    int n   = idx & (N_PIX - 1);
    int c   = (idx >> 12) & (C_DIM - 1);
    int b   = idx >> 19;

    const float* xp = x + (size_t)b * C_DIM * N_PIX + n;
    const float* wp = wv + c * C_DIM;
    float s = bv[c];
#pragma unroll 8
    for (int cc = 0; cc < C_DIM; ++cc)
        s = fmaf(xp[(size_t)cc * N_PIX], wp[cc], s);
    vout[((size_t)b * C_DIM + c) * N_PIX + n] = (bf16)s;
}

// ------------------------------ flash attention ----------------------------
// Two waves per block: wave w streams m in [w*2048, (w+1)*2048) in tiles of
// 64 for one 16-row n-tile; partial flash states merged through LDS.
__global__ __launch_bounds__(64)
void flash_attn_kernel(const bf16* __restrict__ qT,    // [B][N][32]
                       const bf16* __restrict__ kT,    // [B][N][32]
                       const bf16* __restrict__ vbf,   // [B][C][N]
                       const float* __restrict__ x,    // [B][C][N]
                       const float* __restrict__ gamma,
                       float* __restrict__ out)        // [B][C][N]
{
    const int lane = threadIdx.x & 31;  // lane within wave
    const int wv   = threadIdx.x >> 5;  // wave id: m-range selector
    const int l16  = lane & 15;
    const int hh   = lane >> 4;         // wave half: selects K sub-range
    const int n0   = blockIdx.x * 16;
    const int b    = blockIdx.y;

    __shared__ bf16  ldsP[2][16 * MT];  // per-wave exponentiated score tile
    __shared__ float xAcc[8][32 * 8];   // wave-1 acc tiles   (8 KB)
    __shared__ float xLM[2][32 * 8];    // wave-1 lsum / mrow (2 KB)

    // ---- Q as WMMA A operand (16x32, K = d padded to 32) ----
    const bf16* qrow = qT + ((size_t)b * N_PIX + n0 + l16) * 32;
    v8bf qlo = *(const v8bf*)(qrow + hh * 16);       // K 0..7  / 16..23 (=0)
    v8bf qhi = *(const v8bf*)(qrow + hh * 16 + 8);   // K 8..15 / 24..31 (=0)
    v16bf aq;
#pragma unroll
    for (int i = 0; i < 8; ++i) { aq[i] = qlo[i]; aq[8 + i] = qhi[i]; }

    // ---- all-ones B matrix: P x ones = per-row sums, no shuffles ----
    v16bf vones;
#pragma unroll
    for (int i = 0; i < 16; ++i) vones[i] = (bf16)1.0f;

    v8f acc[8];                         // 8 c-tiles of the 16x128 output slab
#pragma unroll
    for (int t = 0; t < 8; ++t) acc[t] = (v8f){0.f,0.f,0.f,0.f,0.f,0.f,0.f,0.f};
    v8f lsum = (v8f){0.f,0.f,0.f,0.f,0.f,0.f,0.f,0.f};   // running row sums
    float mrow[8];
#pragma unroll
    for (int j = 0; j < 8; ++j) mrow[j] = -1e30f;

    const bf16* kbase = kT  + (size_t)b * N_PIX * 32;
    const bf16* vbase = vbf + (size_t)b * C_DIM * N_PIX;
    bf16* ldsPw = &ldsP[wv][0];

    const int mstart = wv * MHALF;
    const int mend   = mstart + MHALF;

    // ---- prologue: load K tiles for first m-tile (register double buffer) --
    v16bf bk0, bk1, bk2, bk3;
    {
        const bf16* kp = kbase + (size_t)(mstart + l16) * 32 + hh * 16;
        bk0 = *(const v16bf*)(kp);
        bk1 = *(const v16bf*)(kp + 16 * 32);
        bk2 = *(const v16bf*)(kp + 32 * 32);
        bk3 = *(const v16bf*)(kp + 48 * 32);
    }

    for (int m0 = mstart; m0 < mend; m0 += MT) {
        // ---- scores for current K tiles ----
        v8f s[4];
        {
            v8f z = (v8f){0.f,0.f,0.f,0.f,0.f,0.f,0.f,0.f};
            s[0] = __builtin_amdgcn_wmma_f32_16x16x32_bf16(false, aq, false, bk0,
                                                           (short)0, z, false, false);
            s[1] = __builtin_amdgcn_wmma_f32_16x16x32_bf16(false, aq, false, bk1,
                                                           (short)0, z, false, false);
            s[2] = __builtin_amdgcn_wmma_f32_16x16x32_bf16(false, aq, false, bk2,
                                                           (short)0, z, false, false);
            s[3] = __builtin_amdgcn_wmma_f32_16x16x32_bf16(false, aq, false, bk3,
                                                           (short)0, z, false, false);
        }

        // ---- issue next iteration's K loads; they overlap softmax + PV ----
        if (m0 + MT < mend) {
            const bf16* kp = kbase + (size_t)(m0 + MT + l16) * 32 + hh * 16;
            bk0 = *(const v16bf*)(kp);
            bk1 = *(const v16bf*)(kp + 16 * 32);
            bk2 = *(const v16bf*)(kp + 32 * 32);
            bk3 = *(const v16bf*)(kp + 48 * 32);
        }

        // ---- batched row-max butterfly (4 waves of 8 bpermutes) ----
        float tl[8];
#pragma unroll
        for (int j = 0; j < 8; ++j)
            tl[j] = fmaxf(fmaxf(s[0][j], s[1][j]), fmaxf(s[2][j], s[3][j]));
#pragma unroll
        for (int st = 1; st <= 8; st <<= 1) {
            float ot[8];
#pragma unroll
            for (int j = 0; j < 8; ++j) ot[j] = __shfl_xor(tl[j], st, 32);
#pragma unroll
            for (int j = 0; j < 8; ++j) tl[j] = fmaxf(tl[j], ot[j]);
        }

        // ---- exp2 (log2-domain scores), rescale accumulators ----
#pragma unroll
        for (int j = 0; j < 8; ++j) {
            float mnew  = fmaxf(mrow[j], tl[j]);
            float alpha = __builtin_amdgcn_exp2f(mrow[j] - mnew);
            mrow[j] = mnew;
#pragma unroll
            for (int t = 0; t < 4; ++t)
                s[t][j] = __builtin_amdgcn_exp2f(s[t][j] - mnew);
            lsum[j] *= alpha;
#pragma unroll
            for (int t = 0; t < 8; ++t) acc[t][j] *= alpha;
        }

        // ---- bounce P through LDS to reach A-operand layout ----
#pragma unroll
        for (int j = 0; j < 8; ++j) {
            int row = j + hh * 8;                 // C/D layout row
#pragma unroll
            for (int t = 0; t < 4; ++t)
                ldsPw[row * MT + 16 * t + l16] = (bf16)s[t][j];
        }
        asm volatile("s_wait_dscnt 0" ::: "memory");

        const bf16* prow = &ldsPw[l16 * MT];
        v16bf ap0, ap1;                           // P cols 0..31 / 32..63
        {
            v8bf lo0 = *(const v8bf*)(prow + hh * 16);
            v8bf hi0 = *(const v8bf*)(prow + hh * 16 + 8);
            v8bf lo1 = *(const v8bf*)(prow + 32 + hh * 16);
            v8bf hi1 = *(const v8bf*)(prow + 40 + hh * 16);
#pragma unroll
            for (int i = 0; i < 8; ++i) {
                ap0[i] = lo0[i]; ap0[8 + i] = hi0[i];
                ap1[i] = lo1[i]; ap1[8 + i] = hi1[i];
            }
        }

        // ---- kick off first V tiles of both phases, then do lsum WMMAs ----
        const bf16* vrow0 = vbase + (size_t)l16 * N_PIX + m0 + hh * 16;
        v16bf bvc  = *(const v16bf*)(vrow0);        // phase-1 tile (K lo half)
        v16bf bvf1 = *(const v16bf*)(vrow0 + 32);   // phase-2 first tile (K hi)

        // running row sums: lsum += P x ones (independent of V loads)
        lsum = __builtin_amdgcn_wmma_f32_16x16x32_bf16(false, ap0, false, vones,
                                                       (short)0, lsum, false, false);
        lsum = __builtin_amdgcn_wmma_f32_16x16x32_bf16(false, ap1, false, vones,
                                                       (short)0, lsum, false, false);

        // ---- phase 1: acc[ct] += P(:,0:32) x V(0:32, ct) — pipelined ----
#pragma unroll
        for (int ct = 0; ct < 8; ++ct) {
            v16bf nb;
            if (ct < 7)
                nb = *(const v16bf*)(vbase + (size_t)((ct + 1) * 16 + l16) * N_PIX
                                     + m0 + hh * 16);
            acc[ct] = __builtin_amdgcn_wmma_f32_16x16x32_bf16(false, ap0, false,
                                                              bvc, (short)0,
                                                              acc[ct], false, false);
            if (ct < 7) bvc = nb;
        }

        // ---- phase 2: acc[ct] += P(:,32:64) x V(32:64, ct) — pipelined ----
        bvc = bvf1;
#pragma unroll
        for (int ct = 0; ct < 8; ++ct) {
            v16bf nb;
            if (ct < 7)
                nb = *(const v16bf*)(vbase + (size_t)((ct + 1) * 16 + l16) * N_PIX
                                     + m0 + 32 + hh * 16);
            acc[ct] = __builtin_amdgcn_wmma_f32_16x16x32_bf16(false, ap1, false,
                                                              bvc, (short)0,
                                                              acc[ct], false, false);
            if (ct < 7) bvc = nb;
        }
    }

    // ---- wave 1 exports its partial flash state to LDS ----
    if (wv == 1) {
#pragma unroll
        for (int ct = 0; ct < 8; ++ct)
            *(v8f*)&xAcc[ct][lane * 8] = acc[ct];
        v8f lv, mv;
#pragma unroll
        for (int j = 0; j < 8; ++j) { lv[j] = lsum[j]; mv[j] = mrow[j]; }
        *(v8f*)&xLM[0][lane * 8] = lv;
        *(v8f*)&xLM[1][lane * 8] = mv;
    }
    __syncthreads();

    // ---- wave 0 merges the two states (element-wise per lane) and writes ---
    if (wv == 0) {
        v8f l1 = *(const v8f*)&xLM[0][lane * 8];
        v8f m1 = *(const v8f*)&xLM[1][lane * 8];
        float sc0[8], sc1[8], rl[8];
#pragma unroll
        for (int j = 0; j < 8; ++j) {
            float M  = fmaxf(mrow[j], m1[j]);
            sc0[j]   = __builtin_amdgcn_exp2f(mrow[j] - M);
            sc1[j]   = __builtin_amdgcn_exp2f(m1[j] - M);
            rl[j]    = 1.0f / (lsum[j] * sc0[j] + l1[j] * sc1[j]);
        }
        const float g = gamma[0];
#pragma unroll
        for (int ct = 0; ct < 8; ++ct) {
            v8f a1 = *(const v8f*)&xAcc[ct][lane * 8];
            int c = ct * 16 + l16;
            const size_t base = ((size_t)b * C_DIM + c) * N_PIX;
#pragma unroll
            for (int j = 0; j < 8; ++j) {
                int n   = n0 + j + hh * 8;
                float o = (acc[ct][j] * sc0[j] + a1[j] * sc1[j]) * rl[j];
                out[base + n] = fmaf(g, o, x[base + n]);
            }
        }
    }
}

// --------------------------------- launch ----------------------------------

extern "C" void kernel_launch(void* const* d_in, const int* in_sizes, int n_in,
                              void* d_out, int out_size, void* d_ws, size_t ws_size,
                              hipStream_t stream) {
    const float* x     = (const float*)d_in[0];
    const float* wq    = (const float*)d_in[1];
    const float* bq    = (const float*)d_in[2];
    const float* wk    = (const float*)d_in[3];
    const float* bk    = (const float*)d_in[4];
    const float* wv    = (const float*)d_in[5];
    const float* bv    = (const float*)d_in[6];
    const float* gamma = (const float*)d_in[7];
    float* out = (float*)d_out;

    bf16* qT = (bf16*)d_ws;                                    // B*N*32 = 512K el
    bf16* kT = qT + (size_t)B_DIM * N_PIX * 32;                // 512K el
    bf16* vb = kT + (size_t)B_DIM * N_PIX * 32;                // B*C*N = 2M el

    // q,k projections: 2 * B * N * 16 = 524288 threads
    proj_qk_kernel<<<2048, 256, 0, stream>>>(x, wq, bq, wk, bk, qT, kT);
    // v projection: B * C * N = 2097152 threads
    proj_v_kernel<<<8192, 256, 0, stream>>>(x, wv, bv, vb);
    // flash attention: 2 waves per 16-row tile (split-m), merged through LDS
    flash_attn_kernel<<<dim3(N_PIX / 16, B_DIM), 64, 0, stream>>>(
        qT, kT, vb, x, gamma, out);
}
